// TransformerBlock_83099027243235
// MI455X (gfx1250) — matrix operand order repaired and checked by reading, not compiled, not run on hardware
//
#include <hip/hip_runtime.h>
#include <hip/hip_bf16.h>

#define B_  8
#define S_  1024
#define D_  1024
#define H_  16
#define F_  4096
#define V_  8192
#define KTOP 64
#define HD_ 64
#define ALPHA_ 0.1f
#define BS_ (B_ * S_)

typedef _Float16 v16h __attribute__((ext_vector_type(16)));
typedef _Float16 v8h  __attribute__((ext_vector_type(8)));
typedef float    v8f  __attribute__((ext_vector_type(8)));

// ---------------------------------------------------------------------------
// WMMA helper (CDNA5 16x16x32 f16 -> f32)
// ---------------------------------------------------------------------------
__device__ __forceinline__ v8f wmma_f16(v16h a, v16h b, v8f c) {
  return __builtin_amdgcn_wmma_f32_16x16x32_f16(
      /*neg_a=*/false, a, /*neg_b=*/false, b,
      /*c_mod=*/(short)0, c, /*reuse_a=*/false, /*reuse_b=*/false);
}

// A-matrix fragment (16x32 f16). Lane L: row = L&15; K-groups {kb..kb+7} and
// {kb+16..kb+23}, kb = (L>=16)?8:0. Two aligned b128 loads.
__device__ __forceinline__ v16h frag_a(const _Float16* p, int stride, int k0) {
  const int lane = threadIdx.x & 31;
  const int m  = lane & 15;
  const int kb = (lane >> 4) << 3;
  const _Float16* q = p + (size_t)m * stride + k0 + kb;
  v8h lo = *(const v8h*)q;
  v8h hi = *(const v8h*)(q + 16);
  v16h f;
#pragma unroll
  for (int e = 0; e < 8; ++e) { f[e] = lo[e]; f[e + 8] = hi[e]; }
  return f;
}

// B-matrix fragment (32x16 f16) from a transposed [N][K] layout.
// Lane L: col = L&15; K-groups {kb..kb+7} and {kb+16..kb+23}, kb = (L>=16)?8:0.
__device__ __forceinline__ v16h frag_b(const _Float16* p, int stride, int k0) {
  const int lane = threadIdx.x & 31;
  const int n  = lane & 15;
  const int kb = (lane >> 4) << 3;
  const _Float16* q = p + (size_t)n * stride + k0 + kb;
  v8h lo = *(const v8h*)q;
  v8h hi = *(const v8h*)(q + 16);
  v16h f;
#pragma unroll
  for (int e = 0; e < 8; ++e) { f[e] = lo[e]; f[e + 8] = hi[e]; }
  return f;
}

// ---------------------------------------------------------------------------
// f32 -> f16 conversion (x)
// ---------------------------------------------------------------------------
__global__ void cvt_kernel(const float* __restrict__ src,
                           _Float16* __restrict__ dst, size_t n) {
  size_t i = (size_t)blockIdx.x * blockDim.x + threadIdx.x;
  size_t st = (size_t)gridDim.x * blockDim.x;
  for (; i < n; i += st) dst[i] = (_Float16)src[i];
}

// ---------------------------------------------------------------------------
// One-time weight convert + transpose: src[K][N] f32 -> dst[N][K] f16.
// Amortized over M=8192 rows; makes GEMM B-tile staging contiguous b128s.
// ---------------------------------------------------------------------------
__global__ void cvtT_kernel(const float* __restrict__ src,
                            _Float16* __restrict__ dst, int Kd, int N) {
  size_t total = (size_t)Kd * N;
  size_t i = (size_t)blockIdx.x * blockDim.x + threadIdx.x;
  size_t st = (size_t)gridDim.x * blockDim.x;
  for (; i < total; i += st) {
    int n = (int)(i / Kd);
    int k = (int)(i % Kd);
    dst[i] = (_Float16)src[(size_t)k * N + n];
  }
}

// ---------------------------------------------------------------------------
// V transpose: V[b,s,h*HD+hd] -> Vt[((b*H+h)*HD+hd)*S + s]
// ---------------------------------------------------------------------------
__global__ void vtrans_kernel(const _Float16* __restrict__ Vm,
                              _Float16* __restrict__ Vt) {
  size_t n = (size_t)B_ * H_ * HD_ * S_;
  size_t i = (size_t)blockIdx.x * blockDim.x + threadIdx.x;
  size_t st = (size_t)gridDim.x * blockDim.x;
  for (; i < n; i += st) {
    int s = (int)(i % S_);
    size_t r = i / S_;
    int hd = (int)(r % HD_); r /= HD_;
    int h  = (int)(r % H_);
    int b  = (int)(r / H_);
    Vt[i] = Vm[((size_t)(b * S_ + s)) * D_ + h * HD_ + hd];
  }
}

// ---------------------------------------------------------------------------
// Tiled WMMA GEMM: C[M,N] = act(A[M,K] @ Bt[N,K]^T + bias)
//  * block tile 128x128, 8 waves, each wave: 16 rows x 128 cols (8 accums)
//  * K-step 32, double-buffered LDS, software-pipelined global loads
// ---------------------------------------------------------------------------
__global__ __launch_bounds__(256) void gemm_kernel(
    const _Float16* __restrict__ A, const _Float16* __restrict__ Bt,
    const float* __restrict__ bias, float* __restrict__ Cf,
    _Float16* __restrict__ Ch, int M, int N, int Kd, int act) {
  __shared__ _Float16 Asl[2][128 * 32];   // 2 x 8 KB
  __shared__ _Float16 Btl[2][128 * 32];   // 2 x 8 KB
  const int m0 = blockIdx.y * 128;
  const int n0 = blockIdx.x * 128;
  const int t  = threadIdx.x;
  const int wv = t >> 5;
  const int lane = t & 31;
  const v8f vzero = {0.f, 0.f, 0.f, 0.f, 0.f, 0.f, 0.f, 0.f};
  v8f acc[8] = {vzero, vzero, vzero, vzero, vzero, vzero, vzero, vzero};

  // staging map: each thread moves 16 halves of A and 16 halves of Bt
  const int sr = t >> 1;                 // row (A) / col (Bt) 0..127
  const int sk = (t & 1) * 16;           // k offset 0 or 16
  const _Float16* arow = A  + (size_t)(m0 + sr) * Kd + sk;
  const _Float16* brow = Bt + (size_t)(n0 + sr) * Kd + sk;

  // preload tile 0
  {
    v8h a0 = *(const v8h*)(arow);
    v8h a1 = *(const v8h*)(arow + 8);
    v8h b0 = *(const v8h*)(brow);
    v8h b1 = *(const v8h*)(brow + 8);
    *(v8h*)(Asl[0] + sr * 32 + sk)     = a0;
    *(v8h*)(Asl[0] + sr * 32 + sk + 8) = a1;
    *(v8h*)(Btl[0] + sr * 32 + sk)     = b0;
    *(v8h*)(Btl[0] + sr * 32 + sk + 8) = b1;
  }
  __syncthreads();

  for (int kt = 0; kt < Kd; kt += 32) {
    const int cur = (kt >> 5) & 1;
    const bool hasNext = (kt + 32) < Kd;
    v8h na0, na1, nb0, nb1;
    if (hasNext) {                       // issue next-tile loads first
      na0 = *(const v8h*)(arow + kt + 32);
      na1 = *(const v8h*)(arow + kt + 40);
      nb0 = *(const v8h*)(brow + kt + 32);
      nb1 = *(const v8h*)(brow + kt + 40);
      if (kt + 64 < Kd) __builtin_prefetch(arow + kt + 64, 0, 1);
    }
    // compute on current tile while loads are in flight
    v16h a = frag_a(Asl[cur] + wv * 16 * 32, 32, 0);
#pragma unroll
    for (int nb = 0; nb < 8; ++nb) {
      v16h bf = frag_b(Btl[cur] + nb * 16 * 32, 32, 0);
      acc[nb] = wmma_f16(a, bf, acc[nb]);
    }
    if (hasNext) {                       // commit next tile to the other buffer
      const int nxt = cur ^ 1;
      *(v8h*)(Asl[nxt] + sr * 32 + sk)     = na0;
      *(v8h*)(Asl[nxt] + sr * 32 + sk + 8) = na1;
      *(v8h*)(Btl[nxt] + sr * 32 + sk)     = nb0;
      *(v8h*)(Btl[nxt] + sr * 32 + sk + 8) = nb1;
    }
    __syncthreads();
  }

  const int mrow = (lane >> 4) << 3;
  const int ncol = lane & 15;
#pragma unroll
  for (int nb = 0; nb < 8; ++nb) {
    int n = n0 + nb * 16 + ncol;
    float bi = bias ? bias[n] : 0.f;
#pragma unroll
    for (int r = 0; r < 8; ++r) {
      int m = m0 + wv * 16 + r + mrow;
      float v = acc[nb][r] + bi;
      if (act == 1) v = 0.5f * v * (1.f + erff(v * 0.70710678118654752f));
      size_t o = (size_t)m * N + n;
      if (Cf) Cf[o] = v;
      if (Ch) Ch[o] = (_Float16)v;
    }
  }
}

// ---------------------------------------------------------------------------
// Attention: one wave per 16-row query tile per (b,h).
// scores -> LDS (f32, 64KB), fused pmi-bias + radix-select top-64 + masked
// softmax, f16 weights written in-place over consumed scores, then w@V WMMA.
// ---------------------------------------------------------------------------
__global__ __launch_bounds__(32) void attention_kernel(
    const _Float16* __restrict__ Q, const _Float16* __restrict__ Km,
    const _Float16* __restrict__ Vt, const int* __restrict__ tok,
    const float* __restrict__ pmi, _Float16* __restrict__ attnh) {
  __shared__ float sc[16 * 1024];                // 64 KB
  _Float16* wlds = (_Float16*)sc;                // overlay (first 32 KB)
  const int it = blockIdx.x, h = blockIdx.y, b = blockIdx.z;
  const int i0 = it * 16;
  const int lane = threadIdx.x;
  const int mrow = (lane >> 4) << 3;
  const int ncol = lane & 15;
  const v8f vzero = {0.f, 0.f, 0.f, 0.f, 0.f, 0.f, 0.f, 0.f};

  const _Float16* qbase = Q + ((size_t)(b * S_ + i0)) * D_ + h * HD_;
  v16h qa0 = frag_a(qbase, D_, 0);
  v16h qa1 = frag_a(qbase, D_, 32);

  for (int jt = 0; jt < S_ / 16; ++jt) {
    const int j0 = jt * 16;
    const _Float16* kbase = Km + ((size_t)(b * S_ + j0)) * D_ + h * HD_;
    v16h kb0 = frag_b(kbase, D_, 0);
    v16h kb1 = frag_b(kbase, D_, 32);
    v8f acc = vzero;
    acc = wmma_f16(qa0, kb0, acc);
    acc = wmma_f16(qa1, kb1, acc);
#pragma unroll
    for (int r = 0; r < 8; ++r)
      sc[(r + mrow) * 1024 + j0 + ncol] = acc[r] * 0.125f;   // 1/sqrt(64)
  }
  __syncthreads();

  const int* tb = tok + b * S_;
  for (int rr = 0; rr < 16; ++rr) {
    const int ti = tb[i0 + rr];
    const float* prow = pmi + (size_t)ti * V_;
    float s[32];
#pragma unroll
    for (int t = 0; t < 32; ++t) {
      int j = lane + 32 * t;
      s[t] = sc[rr * 1024 + j] + ALPHA_ * prow[tb[j]];
    }
    unsigned u[32];
#pragma unroll
    for (int t = 0; t < 32; ++t) {
      unsigned bits = __float_as_uint(s[t]);
      u[t] = (bits & 0x80000000u) ? ~bits : (bits | 0x80000000u);
    }
    // radix-select: largest thr with count(u >= thr) >= KTOP == 64th largest
    unsigned thr = 0u;
    for (int bit = 31; bit >= 0; --bit) {
      unsigned cand = thr | (1u << bit);
      int c = 0;
#pragma unroll
      for (int t = 0; t < 32; ++t) c += (u[t] >= cand) ? 1 : 0;
#pragma unroll
      for (int off = 16; off > 0; off >>= 1) c += __shfl_xor(c, off, 32);
      if (c >= KTOP) thr = cand;
    }
    unsigned kbits = (thr & 0x80000000u) ? (thr & 0x7FFFFFFFu) : ~thr;
    float kth = __uint_as_float(kbits);
    float mx = -3.402823466e38f;
#pragma unroll
    for (int t = 0; t < 32; ++t) mx = fmaxf(mx, s[t]);
#pragma unroll
    for (int off = 16; off > 0; off >>= 1) mx = fmaxf(mx, __shfl_xor(mx, off, 32));
    float sum = 0.f;
#pragma unroll
    for (int t = 0; t < 32; ++t) {
      float e = (s[t] >= kth) ? __expf(s[t] - mx) : 0.f;
      s[t] = e; sum += e;
    }
#pragma unroll
    for (int off = 16; off > 0; off >>= 1) sum += __shfl_xor(sum, off, 32);
    float inv = 1.f / sum;
#pragma unroll
    for (int t = 0; t < 32; ++t)
      wlds[rr * 1024 + lane + 32 * t] = (_Float16)(s[t] * inv);
  }
  __syncthreads();

  v8f acc[4] = {vzero, vzero, vzero, vzero};
  const _Float16* vb = Vt + ((size_t)(b * H_ + h) * HD_) * S_;
  for (int ks = 0; ks < 32; ++ks) {
    v16h a = frag_a(wlds, 1024, ks * 32);
#pragma unroll
    for (int nb = 0; nb < 4; ++nb) {
      v16h bf = frag_b(vb + (size_t)(nb * 16) * S_, S_, ks * 32);
      acc[nb] = wmma_f16(a, bf, acc[nb]);
    }
  }
#pragma unroll
  for (int nb = 0; nb < 4; ++nb) {
#pragma unroll
    for (int r = 0; r < 8; ++r) {
      int m = i0 + r + mrow;
      attnh[((size_t)(b * S_ + m)) * D_ + h * HD_ + nb * 16 + ncol] =
          (_Float16)acc[nb][r];
    }
  }
}

// ---------------------------------------------------------------------------
// out = layernorm(xa + xb) * g + be ; wave per row (D=1024 -> 32 elems/lane)
// ---------------------------------------------------------------------------
__global__ __launch_bounds__(256) void resid_ln_kernel(
    const float* __restrict__ xa, const float* __restrict__ xb,
    const float* __restrict__ g, const float* __restrict__ be,
    float* __restrict__ outf, _Float16* __restrict__ outh, int rows) {
  int row = blockIdx.x * 8 + (threadIdx.x >> 5);
  if (row >= rows) return;
  int lane = threadIdx.x & 31;
  const float* pa = xa + (size_t)row * D_;
  const float* pb = xb + (size_t)row * D_;
  float v[32];
  float sum = 0.f, sq = 0.f;
#pragma unroll
  for (int t = 0; t < 32; ++t) {
    int i = lane + 32 * t;
    float x = pa[i] + pb[i];
    v[t] = x; sum += x; sq += x * x;
  }
#pragma unroll
  for (int off = 16; off > 0; off >>= 1) {
    sum += __shfl_xor(sum, off, 32);
    sq  += __shfl_xor(sq,  off, 32);
  }
  float mu  = sum * (1.f / D_);
  float var = sq * (1.f / D_) - mu * mu;
  float rs  = rsqrtf(var + 1e-5f);
#pragma unroll
  for (int t = 0; t < 32; ++t) {
    int i = lane + 32 * t;
    float o = (v[t] - mu) * rs * g[i] + be[i];
    if (outf) outf[(size_t)row * D_ + i] = o;
    if (outh) outh[(size_t)row * D_ + i] = (_Float16)o;
  }
}

// ---------------------------------------------------------------------------
// Host-side orchestration
// ---------------------------------------------------------------------------
extern "C" void kernel_launch(void* const* d_in, const int* in_sizes, int n_in,
                              void* d_out, int out_size, void* d_ws,
                              size_t ws_size, hipStream_t stream) {
  (void)in_sizes; (void)n_in; (void)out_size; (void)ws_size;
  const float* x   = (const float*)d_in[0];
  const int*   tok = (const int*)d_in[1];
  const float* pmi = (const float*)d_in[2];
  const float* Wq  = (const float*)d_in[3];  const float* bq  = (const float*)d_in[4];
  const float* Wk  = (const float*)d_in[5];  const float* bk  = (const float*)d_in[6];
  const float* Wv  = (const float*)d_in[7];  const float* bv  = (const float*)d_in[8];
  const float* Wo  = (const float*)d_in[9];  const float* bo  = (const float*)d_in[10];
  const float* W1  = (const float*)d_in[11]; const float* b1  = (const float*)d_in[12];
  const float* W2  = (const float*)d_in[13]; const float* b2  = (const float*)d_in[14];
  const float* g1  = (const float*)d_in[15]; const float* be1 = (const float*)d_in[16];
  const float* g2  = (const float*)d_in[17]; const float* be2 = (const float*)d_in[18];

  char* ws = (char*)d_ws;
  size_t off = 0;
  auto alloc = [&](size_t bytes) -> void* {
    void* p = ws + off;
    off += (bytes + 255) & ~(size_t)255;
    return p;
  };

  _Float16* xh   = (_Float16*)alloc((size_t)BS_ * D_ * 2);
  _Float16* WqT  = (_Float16*)alloc((size_t)D_ * D_ * 2);   // [N][K]
  _Float16* WkT  = (_Float16*)alloc((size_t)D_ * D_ * 2);
  _Float16* WvT  = (_Float16*)alloc((size_t)D_ * D_ * 2);
  _Float16* WoT  = (_Float16*)alloc((size_t)D_ * D_ * 2);
  _Float16* W1T  = (_Float16*)alloc((size_t)D_ * F_ * 2);   // [F][D]
  _Float16* W2T  = (_Float16*)alloc((size_t)F_ * D_ * 2);   // [D][F]
  _Float16* Qh   = (_Float16*)alloc((size_t)BS_ * D_ * 2);
  _Float16* Kh   = (_Float16*)alloc((size_t)BS_ * D_ * 2);
  _Float16* Vh   = (_Float16*)alloc((size_t)BS_ * D_ * 2);
  _Float16* Vth  = (_Float16*)alloc((size_t)BS_ * D_ * 2);
  _Float16* ath  = (_Float16*)alloc((size_t)BS_ * D_ * 2);
  float*    atp  = (float*)alloc((size_t)BS_ * D_ * 4);
  float*    hf   = (float*)alloc((size_t)BS_ * D_ * 4);
  _Float16* hh   = (_Float16*)alloc((size_t)BS_ * D_ * 2);
  _Float16* uh   = (_Float16*)alloc((size_t)BS_ * F_ * 2);
  float*    ff   = (float*)alloc((size_t)BS_ * D_ * 4);

  // 1) precision conversion; weights also transposed to [N][K] once
  cvt_kernel <<<4096, 256, 0, stream>>>(x, xh, (size_t)BS_ * D_);
  cvtT_kernel<<<1024, 256, 0, stream>>>(Wq, WqT, D_, D_);
  cvtT_kernel<<<1024, 256, 0, stream>>>(Wk, WkT, D_, D_);
  cvtT_kernel<<<1024, 256, 0, stream>>>(Wv, WvT, D_, D_);
  cvtT_kernel<<<1024, 256, 0, stream>>>(Wo, WoT, D_, D_);
  cvtT_kernel<<<4096, 256, 0, stream>>>(W1, W1T, D_, F_);   // -> [F][D]
  cvtT_kernel<<<4096, 256, 0, stream>>>(W2, W2T, F_, D_);   // -> [D][F]

  // 2) Q/K/V projections
  dim3 gproj(D_ / 128, BS_ / 128);
  gemm_kernel<<<gproj, 256, 0, stream>>>(xh, WqT, bq, nullptr, Qh, BS_, D_, D_, 0);
  gemm_kernel<<<gproj, 256, 0, stream>>>(xh, WkT, bk, nullptr, Kh, BS_, D_, D_, 0);
  gemm_kernel<<<gproj, 256, 0, stream>>>(xh, WvT, bv, nullptr, Vh, BS_, D_, D_, 0);

  // 3) transpose V per (b,h) for contiguous WMMA B-fragments
  vtrans_kernel<<<4096, 256, 0, stream>>>(Vh, Vth);

  // 4) attention (scores + pmi bias + top-64 + softmax + w@V)
  dim3 gattn(S_ / 16, H_, B_);
  attention_kernel<<<gattn, 32, 0, stream>>>(Qh, Kh, Vth, tok, pmi, ath);

  // 5) output projection + residual + LN1
  gemm_kernel<<<gproj, 256, 0, stream>>>(ath, WoT, bo, atp, nullptr, BS_, D_, D_, 0);
  resid_ln_kernel<<<BS_ / 8, 256, 0, stream>>>(x, atp, g1, be1, hf, hh, BS_);

  // 6) FFN (exact GELU fused in first GEMM epilogue) + residual + LN2 -> d_out
  dim3 gff1(F_ / 128, BS_ / 128);
  gemm_kernel<<<gff1, 256, 0, stream>>>(hh, W1T, b1, nullptr, uh, BS_, F_, D_, 1);
  gemm_kernel<<<gproj, 256, 0, stream>>>(uh, W2T, b2, ff, nullptr, BS_, D_, F_, 0);
  resid_ln_kernel<<<BS_ / 8, 256, 0, stream>>>(hf, ff, g2, be2, (float*)d_out,
                                               nullptr, BS_);
}
